// LIFLayer_28613072126564
// MI455X (gfx1250) — matrix operand order repaired and checked by reading, not compiled, not run on hardware
//
#include <hip/hip_runtime.h>
#include <hip/hip_bf16.h>

typedef __bf16 bf16_t;
typedef __attribute__((ext_vector_type(16))) __bf16 v16bf;
typedef __attribute__((ext_vector_type(8)))  __bf16 v8bf;
typedef __attribute__((ext_vector_type(2)))  __bf16 v2bf;
typedef __attribute__((ext_vector_type(8)))  float  v8f;

#define B_  64
#define T_  1000
#define I_  1024
#define H_  1024
#define M_  (B_ * T_)          // 64000 rows, divisible by 128

#define BM  128
#define BN  128
#define BK  32
#define AP  40                 // padded A LDS row (BK+8): 80B stride, 16B-aligned chunks
#define BP  136                // padded B LDS row (BN+8): 272B stride, 16B-aligned chunks

#define ALPHA_MIN_F 0.8187307530779818f
#define ALPHA_MAX_F 0.9607894391523232f

// low 32 bits of a flat LDS pointer == LDS byte offset (ISA 10.2 aperture rule)
__device__ __forceinline__ uint32_t lds_off(const void* p)
{
    return (uint32_t)(uintptr_t)p;
}

// async DMA: 16 bytes global -> LDS, tracked by ASYNCcnt (no VGPR data)
__device__ __forceinline__ void async_b128(uint32_t lds, uint64_t gaddr)
{
    asm volatile("global_load_async_to_lds_b128 %0, %1, off"
                 :: "v"(lds), "v"(gaddr) : "memory");
}

__device__ __forceinline__ void wait_async0()
{
    asm volatile("s_wait_asynccnt 0x0" ::: "memory");
}

// ---------------------------------------------------------------------------
// Kernel 0: X fp32 -> Xb bf16 (8 elems/thread, b128 in / b128 out)
// ---------------------------------------------------------------------------
__global__ __launch_bounds__(256)
void lif_xconv_kernel(const float* __restrict__ X, bf16_t* __restrict__ Xb)
{
    size_t i = ((size_t)blockIdx.x * 256 + threadIdx.x) * 8;
    float4 a = *(const float4*)&X[i];
    float4 b = *(const float4*)&X[i + 4];
    v8bf v;
    v[0] = (bf16_t)a.x; v[1] = (bf16_t)a.y; v[2] = (bf16_t)a.z; v[3] = (bf16_t)a.w;
    v[4] = (bf16_t)b.x; v[5] = (bf16_t)b.y; v[6] = (bf16_t)b.z; v[7] = (bf16_t)b.w;
    *(v8bf*)&Xb[i] = v;
}

// ---------------------------------------------------------------------------
// Kernel 1: W [H][I] fp32 -> Wt [I][H] bf16  (transpose + convert, LDS tiled)
// ---------------------------------------------------------------------------
__global__ __launch_bounds__(256)
void lif_wt_kernel(const float* __restrict__ W, bf16_t* __restrict__ Wt)
{
    __shared__ float tile[32][33];
    const int tx = threadIdx.x & 31;
    const int ty = threadIdx.x >> 5;
    const int i0 = blockIdx.x * 32;
    const int h0 = blockIdx.y * 32;

    for (int r = ty; r < 32; r += 8)
        tile[r][tx] = W[(size_t)(h0 + r) * I_ + i0 + tx];
    __syncthreads();
    for (int r = ty; r < 32; r += 8)
        Wt[(size_t)(i0 + r) * H_ + h0 + tx] = (bf16_t)tile[tx][r];
}

// ---------------------------------------------------------------------------
// Kernel 2: Wx[m][h] = sum_k Xb[m][k] * Wt[k][h]  (bf16 WMMA, fp32 acc)
// Block 128x128x32, 8 waves, wave tile 32x64 (2x4 accumulators).
// A/B tiles staged via async global->LDS DMA (ASYNCcnt), double buffered.
// ---------------------------------------------------------------------------
__global__ __launch_bounds__(256)
void lif_gemm_kernel(const bf16_t* __restrict__ Xb,
                     const bf16_t* __restrict__ Wt,
                     bf16_t* __restrict__ Y)
{
    __shared__ bf16_t As[2][BM][AP];   // 20480 B
    __shared__ bf16_t Bs[2][BK][BP];   // 17408 B

    const int tid   = threadIdx.x;
    const int lane  = tid & 31;
    const int wave  = tid >> 5;
    const int lr    = lane & 15;
    const int lh    = lane >> 4;
    const int waveM = wave & 3;        // 4 slices of 32 rows
    const int waveN = wave >> 2;       // 2 slices of 64 cols

    const int m0 = blockIdx.x * BM;
    const int n0 = blockIdx.y * BN;

    // ---- per-thread async-copy assignments (2 chunks of 16B each for A and B)
    // A tile: 128 rows x 32 bf16 (64B) = 512 chunks; chunk c: rA=c>>2, oA=c&3
    // B tile:  32 rows x 128 bf16 (256B) = 512 chunks; chunk c: rB=c>>4, oB=c&15
    const int cA0 = tid,      cA1 = tid + 256;
    const int rA0 = cA0 >> 2, oA0 = cA0 & 3;
    const int rA1 = cA1 >> 2, oA1 = cA1 & 3;
    const int rB0 = cA0 >> 4, oB0 = cA0 & 15;
    const int rB1 = cA1 >> 4, oB1 = cA1 & 15;

    uint64_t gA0 = (uint64_t)(Xb + (size_t)(m0 + rA0) * I_ + oA0 * 8);
    uint64_t gA1 = (uint64_t)(Xb + (size_t)(m0 + rA1) * I_ + oA1 * 8);
    uint64_t gB0 = (uint64_t)(Wt + (size_t)rB0 * H_ + n0 + oB0 * 8);
    uint64_t gB1 = (uint64_t)(Wt + (size_t)rB1 * H_ + n0 + oB1 * 8);

    uint32_t lA0[2], lA1[2], lB0[2], lB1[2];
#pragma unroll
    for (int b = 0; b < 2; ++b) {
        lA0[b] = lds_off(&As[b][rA0][oA0 * 8]);
        lA1[b] = lds_off(&As[b][rA1][oA1 * 8]);
        lB0[b] = lds_off(&Bs[b][rB0][oB0 * 8]);
        lB1[b] = lds_off(&Bs[b][rB1][oB1 * 8]);
    }

    auto issue_tile = [&](int buf) {
        async_b128(lA0[buf], gA0);
        async_b128(lA1[buf], gA1);
        async_b128(lB0[buf], gB0);
        async_b128(lB1[buf], gB1);
        gA0 += BK * 2;            // +64 B along K
        gA1 += BK * 2;
        gB0 += (size_t)BK * H_ * 2;   // +32 rows
        gB1 += (size_t)BK * H_ * 2;
    };

    v8f acc[2][4];
#pragma unroll
    for (int mi = 0; mi < 2; ++mi)
#pragma unroll
        for (int ni = 0; ni < 4; ++ni)
#pragma unroll
            for (int g = 0; g < 8; ++g)
                acc[mi][ni][g] = 0.0f;

    // prologue: tile 0 into buffer 0
    issue_tile(0);
    wait_async0();
    __syncthreads();

    const int nk = I_ / BK;   // 32 k-steps
    for (int kb = 0; kb < nk; ++kb) {
        const int buf = kb & 1;
        if (kb + 1 < nk) issue_tile(buf ^ 1);   // DMA next tile, overlapped

        // ---- fragments from LDS per ISA 16-bit A(16x32)/B(32x16) layouts ----
        v16bf aF[2], bF[4];
#pragma unroll
        for (int mi = 0; mi < 2; ++mi) {
            const bf16_t* ap = &As[buf][waveM * 32 + mi * 16 + lr][lh * 8];
            v8bf lo = *(const v8bf*)ap;          // K = lh*8 + 0..7
            v8bf hi = *(const v8bf*)(ap + 16);   // K = 16 + lh*8 + 0..7
            aF[mi] = __builtin_shufflevector(lo, hi,
                       0,1,2,3,4,5,6,7,8,9,10,11,12,13,14,15);
        }
#pragma unroll
        for (int ni = 0; ni < 4; ++ni) {
            const bf16_t* bp = &Bs[buf][lh * 8][waveN * 64 + ni * 16 + lr];
#pragma unroll
            for (int s = 0; s < 8; ++s) {
                bF[ni][s]     = bp[s * BP];
                bF[ni][8 + s] = bp[(16 + s) * BP];
            }
        }

#pragma unroll
        for (int mi = 0; mi < 2; ++mi)
#pragma unroll
            for (int ni = 0; ni < 4; ++ni)
                acc[mi][ni] = __builtin_amdgcn_wmma_f32_16x16x32_bf16(
                    false, aF[mi], false, bF[ni],
                    (short)0, acc[mi][ni], false, false);

        if (kb + 1 < nk) wait_async0();         // own DMAs landed in LDS
        __syncthreads();                        // everyone's DMAs landed
    }

    // ---- epilogue: fp32 acc -> bf16 Wx (C layout: elem g -> M=g+8*lh, N=lr)
#pragma unroll
    for (int mi = 0; mi < 2; ++mi) {
#pragma unroll
        for (int ni = 0; ni < 4; ++ni) {
            const int row = m0 + waveM * 32 + mi * 16 + lh * 8;
            const int col = n0 + waveN * 64 + ni * 16 + lr;
#pragma unroll
            for (int g = 0; g < 8; ++g)
                Y[(size_t)(row + g) * H_ + col] = (bf16_t)acc[mi][ni][g];
        }
    }
}

// ---------------------------------------------------------------------------
// Kernel 3: LIF scan over T. Each thread owns 2 adjacent h-chains.
// ---------------------------------------------------------------------------
__global__ __launch_bounds__(256)
void lif_scan_kernel(const bf16_t* __restrict__ Wx,
                     const float* __restrict__ alpha,
                     const float* __restrict__ u0,
                     const float* __restrict__ s0,
                     float* __restrict__ out)
{
    const int idx = blockIdx.x * blockDim.x + threadIdx.x;  // 0..32767
    const int b   = idx / (H_ / 2);
    const int h   = (idx % (H_ / 2)) * 2;

    const float a0 = fminf(fmaxf(alpha[h],     ALPHA_MIN_F), ALPHA_MAX_F);
    const float a1 = fminf(fmaxf(alpha[h + 1], ALPHA_MIN_F), ALPHA_MAX_F);
    const float c0 = 1.0f - a0;
    const float c1 = 1.0f - a1;

    float u_0 = u0[b * H_ + h], u_1 = u0[b * H_ + h + 1];
    float s_0 = s0[b * H_ + h], s_1 = s0[b * H_ + h + 1];

    const bf16_t* wp = Wx  + (size_t)b * T_ * H_ + h;
    float*        op = out + (size_t)b * T_ * H_ + h;

    for (int t = 0; t < T_; ++t) {
        __builtin_prefetch(wp + 8 * H_, 0, 1);
        v2bf w = *(const v2bf*)wp;
        float w0 = (float)w[0], w1 = (float)w[1];

        u_0 = a0 * (u_0 - s_0) + c0 * w0;
        u_1 = a1 * (u_1 - s_1) + c1 * w1;
        s_0 = (u_0 > 1.0f) ? 1.0f : 0.0f;
        s_1 = (u_1 > 1.0f) ? 1.0f : 0.0f;

        float2 o; o.x = s_0; o.y = s_1;
        *(float2*)op = o;

        wp += H_; op += H_;
    }
}

// ---------------------------------------------------------------------------
extern "C" void kernel_launch(void* const* d_in, const int* in_sizes, int n_in,
                              void* d_out, int out_size, void* d_ws, size_t ws_size,
                              hipStream_t stream)
{
    const float* X     = (const float*)d_in[0];   // [64,1000,1024]
    const float* W     = (const float*)d_in[1];   // [1024,1024]
    const float* alpha = (const float*)d_in[2];   // [1024]
    const float* u0    = (const float*)d_in[3];   // [64,1024]
    const float* s0    = (const float*)d_in[4];   // [64,1024]
    float*       out   = (float*)d_out;           // [64,1000,1024]

    bf16_t* Wt = (bf16_t*)d_ws;                       // 1Mi bf16 (2 MB)
    bf16_t* Xb = Wt + (size_t)I_ * H_;                // 64000*1024 bf16 (131 MB)
    bf16_t* Wx = Xb + (size_t)M_ * I_;                // 64000*1024 bf16 (131 MB)

    lif_xconv_kernel<<<(size_t)M_ * I_ / (8 * 256), 256, 0, stream>>>(X, Xb);
    lif_wt_kernel<<<dim3(I_ / 32, H_ / 32), 256, 0, stream>>>(W, Wt);
    lif_gemm_kernel<<<dim3(M_ / BM, H_ / BN), 256, 0, stream>>>(Xb, Wt, Wx);
    lif_scan_kernel<<<(B_ * H_ / 2) / 256, 256, 0, stream>>>(Wx, alpha, u0, s0, out);
}
